// SwinLayer_28673201668297
// MI455X (gfx1250) — matrix operand order repaired and checked by reading, not compiled, not run on hardware
//
#include <hip/hip_runtime.h>
#include <math.h>
#include <stdint.h>

// ---------------------------------------------------------------------------
// Swin layer for MI455X (gfx1250): bf16 WMMA GEMMs (v_wmma_f32_16x16x32_bf16)
// with double-buffered GLOBAL_LOAD_ASYNC_TO_LDS_B128 staging (ASYNCcnt),
// wave32 shuffle softmax, fused bias/residual/GELU epilogues.
// ---------------------------------------------------------------------------

typedef __attribute__((ext_vector_type(16))) __bf16 v16bf;
typedef __attribute__((ext_vector_type(8)))  __bf16 v8bf;
typedef __attribute__((ext_vector_type(8)))  float  v8f;

static constexpr int R   = 65536;   // B * T = 8 * 8192 tokens
static constexpr int C   = 512;
static constexpr int CI  = 2048;    // 4*C
static constexpr int NH_ = 16;
static constexpr int HD  = 32;      // C / NH
static constexpr int WSZ = 128;     // window size
static constexpr int NWIN = R / WSZ; // 512 windows total

__device__ __forceinline__ v8f vzero8() {
    v8f z;
#pragma unroll
    for (int i = 0; i < 8; ++i) z[i] = 0.0f;
    return z;
}

__device__ __forceinline__ v8f wmma_bf16(v16bf a, v16bf b, v8f c) {
    return __builtin_amdgcn_wmma_f32_16x16x32_bf16(
        /*neg_a=*/false, a, /*neg_b=*/false, b,
        /*c_mod=*/(short)0, c, /*reuse_a=*/false, /*reuse_b=*/false);
}

// A-matrix 16x32 bf16 fragment (ISA layout: lane half picks K chunks
// {half*8 .. +8} and {16+half*8 .. +8}); rowp points at K-offset 0 of the row.
__device__ __forceinline__ v16bf load_a_frag(const __bf16* rowp, int half) {
    v8bf lo = *(const v8bf*)(rowp + half * 8);
    v8bf hi = *(const v8bf*)(rowp + 16 + half * 8);
    v16bf r;
#pragma unroll
    for (int i = 0; i < 8; ++i) { r[i] = lo[i]; r[i + 8] = hi[i]; }
    return r;
}

// B-matrix 32x16 bf16 fragment: same K chunks as A ({half*8 .. +8}, {16+half*8 .. +8}).
__device__ __forceinline__ v16bf load_b_frag(const __bf16* rowp, int half) {
    v8bf lo = *(const v8bf*)(rowp + half * 8);
    v8bf hi = *(const v8bf*)(rowp + 16 + half * 8);
    v16bf r;
#pragma unroll
    for (int i = 0; i < 8; ++i) { r[i] = lo[i]; r[i + 8] = hi[i]; }
    return r;
}

// CDNA5 async copy: global -> LDS, 16 bytes per lane, tracked by ASYNCcnt.
__device__ __forceinline__ void async_ld_b128(uint32_t lds_addr, uint64_t gaddr) {
    asm volatile("global_load_async_to_lds_b128 %0, %1, off"
                 :: "v"(lds_addr), "v"(gaddr) : "memory");
}
__device__ __forceinline__ void wait_async_le4() {
    asm volatile("s_wait_asynccnt 0x4" ::: "memory");
}
__device__ __forceinline__ void wait_async_0() {
    asm volatile("s_wait_asynccnt 0x0" ::: "memory");
}

// ---------------------------------------------------------------------------
// fp32 [K,N] -> bf16 transposed [N,K] weight conversion
// ---------------------------------------------------------------------------
__global__ void wt_convert_kernel(const float* __restrict__ src,
                                  __bf16* __restrict__ dst, int K, int N) {
    int idx = blockIdx.x * 256 + threadIdx.x;
    if (idx >= K * N) return;
    int k = idx / N, n = idx % N;
    dst[(size_t)n * K + k] = (__bf16)src[idx];
}

// ---------------------------------------------------------------------------
// LayerNorm over C=512, fp32 in -> bf16 out. One block (128 threads) per row.
// ---------------------------------------------------------------------------
__global__ __launch_bounds__(128) void layernorm_kernel(
    const float* __restrict__ x, const float* __restrict__ g,
    const float* __restrict__ b, __bf16* __restrict__ out) {
    int row = blockIdx.x;
    int tid = threadIdx.x;
    const float4* xr = (const float4*)(x + (size_t)row * C);
    float4 v = xr[tid];
    float s  = v.x + v.y + v.z + v.w;
    float s2 = v.x * v.x + v.y * v.y + v.z * v.z + v.w * v.w;
    __shared__ float red0[128], red1[128];
    red0[tid] = s; red1[tid] = s2;
    __syncthreads();
#pragma unroll
    for (int st = 64; st > 0; st >>= 1) {
        if (tid < st) { red0[tid] += red0[tid + st]; red1[tid] += red1[tid + st]; }
        __syncthreads();
    }
    float mean = red0[0] * (1.0f / C);
    float var  = red1[0] * (1.0f / C) - mean * mean;
    float inv  = rsqrtf(var + 1e-5f);
    int c = tid * 4;
    __bf16* o = out + (size_t)row * C + c;
    o[0] = (__bf16)((v.x - mean) * inv * g[c + 0] + b[c + 0]);
    o[1] = (__bf16)((v.y - mean) * inv * g[c + 1] + b[c + 1]);
    o[2] = (__bf16)((v.z - mean) * inv * g[c + 2] + b[c + 2]);
    o[3] = (__bf16)((v.w - mean) * inv * g[c + 3] + b[c + 3]);
}

// ---------------------------------------------------------------------------
// GEMM: out[M,N] = A[M,K](bf16) @ W (via WT[N,K] bf16) + bias, with epilogue.
// MODE 0: store bf16.  MODE 1: store f32 + residual.  MODE 2: GELU -> bf16.
// Block tile 128x128, BK=32; 8 waves (2M x 4N); wave tile 64x32 (4x2 accums).
// A/B tiles double-buffered in LDS via global_load_async_to_lds_b128.
// ---------------------------------------------------------------------------
template <int MODE>
__global__ __launch_bounds__(256) void gemm_bf16_kernel(
    const __bf16* __restrict__ A, const __bf16* __restrict__ WT,
    const float* __restrict__ bias, const float* resid,
    void* outp, int M, int K, int N) {
    __shared__ alignas(64) __bf16 tA[2][128 * 32];   // 8 KB per buffer
    __shared__ alignas(64) __bf16 tB[2][128 * 32];

    const int tid  = threadIdx.x;
    const int lane = tid & 31;
    const int w    = tid >> 5;
    const int wm = w >> 2, wn = w & 3;
    const int bm = blockIdx.y * 128, bn = blockIdx.x * 128;
    const int half = lane >> 4, l15 = lane & 15;

    // staging: thread covers tile row (tid>>1), 16-element chunk ((tid&1)*16)
    const int srow = tid >> 1;
    const int scol = (tid & 1) * 16;
    const uint64_t gA = (uint64_t)(uintptr_t)(A  + (size_t)(bm + srow) * K + scol);
    const uint64_t gB = (uint64_t)(uintptr_t)(WT + (size_t)(bn + srow) * K + scol);
    const uint32_t lA[2] = { (uint32_t)(uintptr_t)&tA[0][srow * 32 + scol],
                             (uint32_t)(uintptr_t)&tA[1][srow * 32 + scol] };
    const uint32_t lB[2] = { (uint32_t)(uintptr_t)&tB[0][srow * 32 + scol],
                             (uint32_t)(uintptr_t)&tB[1][srow * 32 + scol] };

    v8f acc[4][2];
#pragma unroll
    for (int mt = 0; mt < 4; ++mt)
#pragma unroll
        for (int nt = 0; nt < 2; ++nt) acc[mt][nt] = vzero8();

    // issue stage 0
    {
        async_ld_b128(lA[0],      gA);
        async_ld_b128(lA[0] + 16, gA + 16);
        async_ld_b128(lB[0],      gB);
        async_ld_b128(lB[0] + 16, gB + 16);
    }

    int buf = 0;
    for (int kk = 0; kk < K; kk += 32) {
        if (kk + 32 < K) {
            uint64_t ofs = (uint64_t)((kk + 32) * 2);
            int nb = buf ^ 1;
            async_ld_b128(lA[nb],      gA + ofs);
            async_ld_b128(lA[nb] + 16, gA + ofs + 16);
            async_ld_b128(lB[nb],      gB + ofs);
            async_ld_b128(lB[nb] + 16, gB + ofs + 16);
            wait_async_le4();   // current buf's 4 transfers complete
        } else {
            wait_async_0();
        }
        __syncthreads();        // whole tile visible to all waves

        v16bf af[4], bfr[2];
#pragma unroll
        for (int mt = 0; mt < 4; ++mt)
            af[mt] = load_a_frag(&tA[buf][(wm * 64 + mt * 16 + l15) * 32], half);
#pragma unroll
        for (int nt = 0; nt < 2; ++nt)
            bfr[nt] = load_b_frag(&tB[buf][(wn * 32 + nt * 16 + l15) * 32], half);
#pragma unroll
        for (int mt = 0; mt < 4; ++mt)
#pragma unroll
            for (int nt = 0; nt < 2; ++nt)
                acc[mt][nt] = wmma_bf16(af[mt], bfr[nt], acc[mt][nt]);

        __syncthreads();        // done reading buf before it is overwritten
        buf ^= 1;
    }

#pragma unroll
    for (int mt = 0; mt < 4; ++mt) {
#pragma unroll
        for (int nt = 0; nt < 2; ++nt) {
            int col = bn + wn * 32 + nt * 16 + l15;
            float bv = bias[col];
            size_t row0 = (size_t)(bm + wm * 64 + mt * 16 + half * 8);
#pragma unroll
            for (int r = 0; r < 8; ++r) {
                size_t off = (row0 + r) * (size_t)N + col;
                float vv = acc[mt][nt][r] + bv;
                if (MODE == 0) {
                    ((__bf16*)outp)[off] = (__bf16)vv;
                } else if (MODE == 1) {
                    ((float*)outp)[off] = vv + resid[off];
                } else {
                    float gv = 0.5f * vv * (1.0f + erff(vv * 0.70710678118654752f));
                    ((__bf16*)outp)[off] = (__bf16)gv;
                }
            }
        }
    }
}

// ---------------------------------------------------------------------------
// Windowed MHSA core: one block per (window, head). 8 waves, 16 q-rows each.
// scores = Q Kᵀ / sqrt(hd) + rel_bias; softmax; ctx = P V.  All WMMA bf16.
// ---------------------------------------------------------------------------
__global__ __launch_bounds__(256) void attn_kernel(
    const __bf16* __restrict__ Qb, const __bf16* __restrict__ Kb,
    const __bf16* __restrict__ Vb, const float* __restrict__ rel_bias,
    __bf16* __restrict__ ctx) {
    const int win = blockIdx.x / NH_;
    const int h   = blockIdx.x % NH_;
    const int rowbase = win * WSZ;
    const int colbase = h * HD;
    const int lane = threadIdx.x & 31;
    const int w    = threadIdx.x >> 5;
    const int half = lane >> 4, l15 = lane & 15;

    __shared__ alignas(32) __bf16 VT[32][144];       // V transposed [d][key]
    __shared__ alignas(32) __bf16 ATT[8][16][128];   // per-wave probabilities

    // Stage V transposed: VT[d][k] = V[k][d]
    for (int idx = threadIdx.x; idx < WSZ * HD; idx += 256) {
        int k = idx >> 5, d = idx & 31;
        VT[d][k] = Vb[(size_t)(rowbase + k) * C + colbase + d];
    }
    __syncthreads();

    // Q fragment (A layout, K = hd = 32 in one WMMA step)
    v16bf qa = load_a_frag(Qb + (size_t)(rowbase + w * 16 + l15) * C + colbase, half);

    // scores: 8 key tiles of 16
    v8f s[8];
#pragma unroll
    for (int t = 0; t < 8; ++t) {
        v16bf kb = load_b_frag(Kb + (size_t)(rowbase + t * 16 + l15) * C + colbase, half);
        s[t] = wmma_bf16(qa, kb, vzero8());
    }

    // scale + relative position bias
    const float scale = 0.17677669529663687f; // 1/sqrt(32)
#pragma unroll
    for (int t = 0; t < 8; ++t)
#pragma unroll
        for (int r = 0; r < 8; ++r) {
            int qi = w * 16 + half * 8 + r;
            int kj = t * 16 + l15;
            s[t][r] = s[t][r] * scale + rel_bias[(qi - kj + WSZ - 1) * NH_ + h];
        }

    // softmax over 128 keys per row; row values live across 8 tiles x 16 lanes
    // of this half-wave -> xor-shuffle reductions with masks 8..1 (wave32).
#pragma unroll
    for (int r = 0; r < 8; ++r) {
        float m = -3.0e38f;
#pragma unroll
        for (int t = 0; t < 8; ++t) m = fmaxf(m, s[t][r]);
#pragma unroll
        for (int sh = 8; sh >= 1; sh >>= 1) m = fmaxf(m, __shfl_xor(m, sh, 32));
        float sum = 0.0f;
#pragma unroll
        for (int t = 0; t < 8; ++t) { float e = __expf(s[t][r] - m); s[t][r] = e; sum += e; }
#pragma unroll
        for (int sh = 8; sh >= 1; sh >>= 1) sum += __shfl_xor(sum, sh, 32);
        float inv = 1.0f / sum;
#pragma unroll
        for (int t = 0; t < 8; ++t) s[t][r] *= inv;
    }

    // C-layout -> A-layout reshape through LDS (bf16)
#pragma unroll
    for (int t = 0; t < 8; ++t)
#pragma unroll
        for (int r = 0; r < 8; ++r)
            ATT[w][half * 8 + r][t * 16 + l15] = (__bf16)s[t][r];

    // ctx = P[16,128] @ V[128,32]: 4 K-steps of 32, 2 N-tiles of 16
    v8f o0 = vzero8(), o1 = vzero8();
#pragma unroll
    for (int ks = 0; ks < 4; ++ks) {
        v16bf pa = load_a_frag(&ATT[w][l15][ks * 32], half);
        v16bf v0 = load_b_frag(&VT[l15][ks * 32], half);
        v16bf v1 = load_b_frag(&VT[16 + l15][ks * 32], half);
        o0 = wmma_bf16(pa, v0, o0);
        o1 = wmma_bf16(pa, v1, o1);
    }

#pragma unroll
    for (int r = 0; r < 8; ++r) {
        size_t row = (size_t)(rowbase + w * 16 + half * 8 + r);
        ctx[row * C + colbase + l15]      = (__bf16)o0[r];
        ctx[row * C + colbase + 16 + l15] = (__bf16)o1[r];
    }
}

// ---------------------------------------------------------------------------
// Workspace layout (bytes)
// ---------------------------------------------------------------------------
static constexpr size_t SZ_WCC  = (size_t)C * C * 2;       // 512 KB
static constexpr size_t OFF_WTQ = 0;
static constexpr size_t OFF_WTK = OFF_WTQ + SZ_WCC;
static constexpr size_t OFF_WTV = OFF_WTK + SZ_WCC;
static constexpr size_t OFF_WTO = OFF_WTV + SZ_WCC;
static constexpr size_t OFF_WT1 = OFF_WTO + SZ_WCC;        // [CI,C] bf16, 2 MB
static constexpr size_t OFF_WT2 = OFF_WT1 + (size_t)CI * C * 2; // [C,CI] bf16, 2 MB
static constexpr size_t OFF_ABUF = (size_t)8 << 20;        // ln1 / ctx / ln2 (64 MB)
static constexpr size_t OFF_QBUF = OFF_ABUF + (size_t)R * C * 2;
static constexpr size_t OFF_KBUF = OFF_QBUF + (size_t)R * C * 2;
static constexpr size_t OFF_VBUF = OFF_KBUF + (size_t)R * C * 2;
static constexpr size_t OFF_MBUF = OFF_QBUF;               // mlp1 out overlays Q/K/V (256 MB)

extern "C" void kernel_launch(void* const* d_in, const int* in_sizes, int n_in,
                              void* d_out, int out_size, void* d_ws, size_t ws_size,
                              hipStream_t stream) {
    const float* hidden   = (const float*)d_in[0];
    const float* ln1_g    = (const float*)d_in[1];
    const float* ln1_b    = (const float*)d_in[2];
    const float* wq       = (const float*)d_in[3];
    const float* bq       = (const float*)d_in[4];
    const float* wk       = (const float*)d_in[5];
    const float* bk       = (const float*)d_in[6];
    const float* wv       = (const float*)d_in[7];
    const float* bv       = (const float*)d_in[8];
    const float* wo       = (const float*)d_in[9];
    const float* bo       = (const float*)d_in[10];
    const float* rel_bias = (const float*)d_in[11];
    const float* ln2_g    = (const float*)d_in[12];
    const float* ln2_b    = (const float*)d_in[13];
    const float* w1       = (const float*)d_in[14];
    const float* b1       = (const float*)d_in[15];
    const float* w2       = (const float*)d_in[16];
    const float* b2       = (const float*)d_in[17];

    char* ws = (char*)d_ws;
    __bf16* WTQ  = (__bf16*)(ws + OFF_WTQ);
    __bf16* WTK  = (__bf16*)(ws + OFF_WTK);
    __bf16* WTV  = (__bf16*)(ws + OFF_WTV);
    __bf16* WTO  = (__bf16*)(ws + OFF_WTO);
    __bf16* WT1  = (__bf16*)(ws + OFF_WT1);
    __bf16* WT2  = (__bf16*)(ws + OFF_WT2);
    __bf16* ABUF = (__bf16*)(ws + OFF_ABUF);
    __bf16* QBUF = (__bf16*)(ws + OFF_QBUF);
    __bf16* KBUF = (__bf16*)(ws + OFF_KBUF);
    __bf16* VBUF = (__bf16*)(ws + OFF_VBUF);
    __bf16* MBUF = (__bf16*)(ws + OFF_MBUF);
    float*  xout = (float*)d_out;

    // 1) weights -> bf16, transposed [N,K]
    wt_convert_kernel<<<(C * C + 255) / 256, 256, 0, stream>>>(wq, WTQ, C, C);
    wt_convert_kernel<<<(C * C + 255) / 256, 256, 0, stream>>>(wk, WTK, C, C);
    wt_convert_kernel<<<(C * C + 255) / 256, 256, 0, stream>>>(wv, WTV, C, C);
    wt_convert_kernel<<<(C * C + 255) / 256, 256, 0, stream>>>(wo, WTO, C, C);
    wt_convert_kernel<<<(C * CI + 255) / 256, 256, 0, stream>>>(w1, WT1, C, CI);
    wt_convert_kernel<<<(CI * C + 255) / 256, 256, 0, stream>>>(w2, WT2, CI, C);

    // 2) LN1 -> bf16
    layernorm_kernel<<<R, 128, 0, stream>>>(hidden, ln1_g, ln1_b, ABUF);

    // 3) QKV projections (WMMA GEMM, bias fused, bf16 out)
    dim3 gCC(C / 128, R / 128);
    gemm_bf16_kernel<0><<<gCC, 256, 0, stream>>>(ABUF, WTQ, bq, nullptr, QBUF, R, C, C);
    gemm_bf16_kernel<0><<<gCC, 256, 0, stream>>>(ABUF, WTK, bk, nullptr, KBUF, R, C, C);
    gemm_bf16_kernel<0><<<gCC, 256, 0, stream>>>(ABUF, WTV, bv, nullptr, VBUF, R, C, C);

    // 4) windowed attention (ctx overwrites ABUF; ln1 output no longer needed)
    attn_kernel<<<NWIN * NH_, 256, 0, stream>>>(QBUF, KBUF, VBUF, rel_bias, ABUF);

    // 5) out projection + residual: x = hidden + ctx@wo + bo  (x lives in d_out)
    gemm_bf16_kernel<1><<<gCC, 256, 0, stream>>>(ABUF, WTO, bo, hidden, xout, R, C, C);

    // 6) LN2 -> bf16
    layernorm_kernel<<<R, 128, 0, stream>>>(xout, ln2_g, ln2_b, ABUF);

    // 7) MLP fc1 + exact GELU -> bf16
    dim3 gCI(CI / 128, R / 128);
    gemm_bf16_kernel<2><<<gCI, 256, 0, stream>>>(ABUF, WT1, b1, nullptr, MBUF, R, C, CI);

    // 8) MLP fc2 + residual (in-place on d_out; element-wise read-then-write)
    gemm_bf16_kernel<1><<<gCC, 256, 0, stream>>>(MBUF, WT2, b2, xout, xout, R, CI, C);
}
